// CG_model_jit_40355512713752
// MI455X (gfx1250) — matrix operand order repaired and checked by reading, not compiled, not run on hardware
//
#include <hip/hip_runtime.h>
#include <hip/hip_bf16.h>
#include <math.h>

// ---------------------------------------------------------------------------
// CDNA5 (gfx1250) implementation of the CG-model step.
// Wave32; 16 rows per wave; hidden 32x32 layers via V_WMMA_F32_16X16X32_F16.
// Fast activations: v_exp/v_log/v_rcp intrinsics (no libm expansions).
// ---------------------------------------------------------------------------

typedef __attribute__((ext_vector_type(16))) _Float16 v16h;
typedef __attribute__((ext_vector_type(8)))  float    v8f;

#define WAVES_PER_BLOCK 4
#define TILE 16

__device__ __forceinline__ float fast_rcp(float x) {
  return __builtin_amdgcn_rcpf(x);
}
__device__ __forceinline__ float siluf(float x) {
  return x * __builtin_amdgcn_rcpf(1.f + __expf(-x));
}
// softplus(x) = max(x,0) + log(1 + exp(-|x|))  (stable, fast v_exp/v_log)
__device__ __forceinline__ float softplusf(float x) {
  return fmaxf(x, 0.f) + __logf(1.f + __expf(-fabsf(x)));
}

// Build the two B-matrix (32x16 f16) fragments for a 32x32 weight, y = h @ W^T.
// B[k][n] = W[n][k]. Layout: lane holds column n = lane&15; lanes 0-15 hold
// K = {0..7, 16..23}, lanes 16-31 hold K = {8..15, 24..31}.
__device__ __forceinline__ void build_bfrag(const float* __restrict__ W1,
                                            bool absw, int lane,
                                            v16h& B0, v16h& B1) {
  const int col = lane & 15;
  const int kbase = (lane >> 4) * 16;
#pragma unroll
  for (int e = 0; e < 16; ++e) {
    const int k = (e < 8) ? ((kbase >> 1) + e) : (16 + (kbase >> 1) + (e - 8));
    float w0 = W1[col * 32 + k];          // output columns 0..15
    float w1 = W1[(col + 16) * 32 + k];   // output columns 16..31
    if (absw) { w0 = fabsf(w0); w1 = fabsf(w1); }
    B0[e] = (_Float16)w0;
    B1[e] = (_Float16)w1;
  }
}

// Evaluate a [DIN -> 32 -> 32 -> 1] MLP on 16 rows per wave.
// Row index = lane & 15 (duplicated across the two half-waves).
// ACT: 0 = SiLU, 1 = softplus. ABSW: monotonic net (|W|, first-layer
// T_SIGNS = [+1,-1] applied to input columns).
// Returns y for row (lane&15), valid in all lanes.
template<int ACT, int DIN, bool ABSW>
__device__ __forceinline__ float mlp_eval(
    float x0, float x1,
    const float* __restrict__ W0, const float* __restrict__ b0,
    const v16h& B0, const v16h& B1,
    const float* __restrict__ b1,
    const float* __restrict__ W2, const float* __restrict__ b2,
    int lane)
{
  const int half = lane >> 4;
  const int col  = lane & 15;

  // Layer 0 computed directly in WMMA A-fragment layout:
  // lanes 0-15: K = {0..7, 16..23}; lanes 16-31: K = {8..15, 24..31}.
  v16h Af;
#pragma unroll
  for (int e = 0; e < 16; ++e) {
    const int k = (e < 8) ? (half * 8 + e) : (16 + half * 8 + (e - 8));
    float w00 = W0[k * DIN];
    if (ABSW) w00 = fabsf(w00);
    float y = fmaf(x0, w00, b0[k]);
    if (DIN == 2) {
      float w01 = W0[k * 2 + 1];
      w01 = ABSW ? -fabsf(w01) : w01;   // T_SIGNS = [1, -1]
      y = fmaf(x1, w01, y);
    }
    const float h = (ACT == 0) ? siluf(y) : softplusf(y);
    Af[e] = (_Float16)h;
  }

  // Layer 1 (32x32) via two WMMA 16x16x32 f16 ops, bias in C.
  const float bias0 = b1[col];
  const float bias1 = b1[col + 16];
  v8f c0, c1;
#pragma unroll
  for (int r = 0; r < 8; ++r) { c0[r] = bias0; c1[r] = bias1; }
  c0 = __builtin_amdgcn_wmma_f32_16x16x32_f16(false, Af, false, B0,
                                              (short)0, c0, false, false);
  c1 = __builtin_amdgcn_wmma_f32_16x16x32_f16(false, Af, false, B1,
                                              (short)0, c1, false, false);

  // Layer 2 (32 -> 1): fold W2 into D-fragment, butterfly-reduce 16 lanes.
  float w2a = W2[col];      if (ABSW) w2a = fabsf(w2a);
  float w2b = W2[col + 16]; if (ABSW) w2b = fabsf(w2b);
  float part[8];
#pragma unroll
  for (int r = 0; r < 8; ++r) {
    const float h0 = (ACT == 0) ? siluf(c0[r]) : softplusf(c0[r]);
    const float h1 = (ACT == 0) ? siluf(c1[r]) : softplusf(c1[r]);
    part[r] = fmaf(h0, w2a, h1 * w2b);
  }
#pragma unroll
  for (int r = 0; r < 8; ++r) {
#pragma unroll
    for (int m = 1; m <= 8; m <<= 1)
      part[r] += __shfl_xor(part[r], m, 32);
  }
  // part[r] (lanes 0-15) = row r ; part[r] (lanes 16-31) = row 8+r.
  float y = b2[0];
#pragma unroll
  for (int r = 0; r < 8; ++r) {
    const float lo = __shfl(part[r], 0, 32);
    const float hi = __shfl(part[r], 16, 32);
    if (col == r)     y += lo;
    if (col == r + 8) y += hi;
  }
  return y;
}

// ---------------------------------------------------------------------------
// Node kernel: monotonic E_mlp at 4 perturbed inputs -> U, T, P, C per node.
// ---------------------------------------------------------------------------
__global__ void node_kernel(const float* __restrict__ S,
                            const float* __restrict__ dArr,
                            const float* eW0, const float* eb0,
                            const float* eW1, const float* eb1,
                            const float* eW2, const float* eb2,
                            float* __restrict__ Tn, float* __restrict__ Pn,
                            float* __restrict__ Cn, float* __restrict__ Un,
                            int N)
{
  const int lane = threadIdx.x & 31;
  const int wv   = threadIdx.x >> 5;
  const int tile = blockIdx.x * WAVES_PER_BLOCK + wv;
  const int row  = lane & 15;
  const int node = tile * TILE + row;
  const int nc   = (node < N) ? node : (N - 1);

  const float s  = S[nc];
  const float Vn = fast_rcp(dArr[nc]);

  v16h B0, B1;
  build_bfrag(eW1, true, lane, B0, B1);

  const float E2 = 0.01f;
  const float U0  = mlp_eval<1,2,true>(s,      Vn,      eW0, eb0, B0, B1, eb1, eW2, eb2, lane);
  const float USp = mlp_eval<1,2,true>(s + E2, Vn,      eW0, eb0, B0, B1, eb1, eW2, eb2, lane);
  const float UVp = mlp_eval<1,2,true>(s,      Vn + E2, eW0, eb0, B0, B1, eb1, eW2, eb2, lane);
  const float USm = mlp_eval<1,2,true>(s - E2, Vn,      eW0, eb0, B0, B1, eb1, eW2, eb2, lane);

  if (node < N && lane < 16) {
    const float T = (USp - U0) * 100.f;          // /E2
    const float P = -(UVp - U0) * 100.f;
    const float C = T * E2 * E2 * fast_rcp(USp - 2.f * U0 + USm);
    Tn[node] = T; Pn[node] = P; Cn[node] = C; Un[node] = U0;
  }
}

// ---------------------------------------------------------------------------
// Edge kernel: 14 MLP evals per 16-edge tile + physics + atomic scatter.
// ---------------------------------------------------------------------------
struct EdgeArgs {
  const float* v;
  const int*   ei;
  const float* rij;
  const float* dArr;
  const float* dWn;
  const float* dV;
  const float *aW0,*ab0,*aW1,*ab1,*aW2,*ab2;
  const float *bW0,*bb0,*bW1,*bb1,*bW2,*bb2;
  const float *cW0,*cb0,*cW1,*cb1,*cW2,*cb2;
  const float *wW0,*wb0,*wW1,*wb1,*wW2,*wb2;
  const float *logm;
  const float *Tn,*Pn,*Cn;
  float *accPd,*accMSV,*accDivV,*accNoise,*accMgSS,*accDivMS,*accTnC;
  int N, E;
};

__global__ void edge_kernel(EdgeArgs a) {
  const int lane = threadIdx.x & 31;
  const int wv   = threadIdx.x >> 5;
  const int tile = blockIdx.x * WAVES_PER_BLOCK + wv;
  const int row  = lane & 15;
  const int e    = tile * TILE + row;
  const bool valid = (e < a.E);
  const int ec = valid ? e : (a.E - 1);

  // Per-wave resident weight fragments (4 MLPs x 2 tiles).
  v16h AB0,AB1,BB0,BB1,CB0,CB1,WB0,WB1;
  build_bfrag(a.aW1, false, lane, AB0, AB1);
  build_bfrag(a.bW1, false, lane, BB0, BB1);
  build_bfrag(a.cW1, false, lane, CB0, CB1);
  build_bfrag(a.wW1, false, lane, WB0, WB1);

  const int i = a.ei[ec];
  const int j = a.ei[a.E + ec];
  const float rx = a.rij[ec*3+0], ry = a.rij[ec*3+1], rz = a.rij[ec*3+2];
  const float rn = sqrtf(rx*rx + ry*ry + rz*rz);
  const float ir = fast_rcp(rn + 1e-8f);
  const float ex = rx*ir, ey = ry*ir, ez = rz*ir;

  const float E1 = 0.001f;

  // Smoothing-kernel gradient via finite difference of W_mlp (H = 1).
  const float Skp = rn + E1, Skm = rn - E1;
  const float yWp = mlp_eval<0,1,false>(Skp, 0.f, a.wW0,a.wb0,WB0,WB1,a.wb1,a.wW2,a.wb2, lane);
  const float yWm = mlp_eval<0,1,false>(Skm, 0.f, a.wW0,a.wb0,WB0,WB1,a.wb1,a.wW2,a.wb2, lane);
  const float Wkp = __expf(yWp) * (1.f - Skp*Skp);
  const float Wkm = __expf(yWm) * (1.f - Skm*Skm);
  const float dWdr = (Wkp - Wkm) * fast_rcp(2.f * E1 * rn);
  const float gWx = dWdr*rx, gWy = dWdr*ry, gWz = dWdr*rz;

  const float Ti = a.Tn[i], Tj = a.Tn[j];
  const float Ci = a.Cn[i], Cj = a.Cn[j];
  const float Pi = a.Pn[i], Pj = a.Pn[j];
  const float di = a.dArr[i], dj = a.dArr[j];

  // A/B/C MLPs at (r_norm, T) for T in {T_i, T_j, T_i+eps, T_j+eps}.
  const float Ami  = mlp_eval<0,2,false>(rn, Ti,    a.aW0,a.ab0,AB0,AB1,a.ab1,a.aW2,a.ab2, lane);
  const float Amj  = mlp_eval<0,2,false>(rn, Tj,    a.aW0,a.ab0,AB0,AB1,a.ab1,a.aW2,a.ab2, lane);
  const float Amie = mlp_eval<0,2,false>(rn, Ti+E1, a.aW0,a.ab0,AB0,AB1,a.ab1,a.aW2,a.ab2, lane);
  const float Amje = mlp_eval<0,2,false>(rn, Tj+E1, a.aW0,a.ab0,AB0,AB1,a.ab1,a.aW2,a.ab2, lane);
  const float Bmi  = mlp_eval<0,2,false>(rn, Ti,    a.bW0,a.bb0,BB0,BB1,a.bb1,a.bW2,a.bb2, lane);
  const float Bmj  = mlp_eval<0,2,false>(rn, Tj,    a.bW0,a.bb0,BB0,BB1,a.bb1,a.bW2,a.bb2, lane);
  const float Bmie = mlp_eval<0,2,false>(rn, Ti+E1, a.bW0,a.bb0,BB0,BB1,a.bb1,a.bW2,a.bb2, lane);
  const float Bmje = mlp_eval<0,2,false>(rn, Tj+E1, a.bW0,a.bb0,BB0,BB1,a.bb1,a.bW2,a.bb2, lane);
  const float Cmi  = mlp_eval<0,2,false>(rn, Ti,    a.cW0,a.cb0,CB0,CB1,a.cb1,a.cW2,a.cb2, lane);
  const float Cmj  = mlp_eval<0,2,false>(rn, Tj,    a.cW0,a.cb0,CB0,CB1,a.cb1,a.cW2,a.cb2, lane);
  const float Cmie = mlp_eval<0,2,false>(rn, Ti+E1, a.cW0,a.cb0,CB0,CB1,a.cb1,a.cW2,a.cb2, lane);
  const float Cmje = mlp_eval<0,2,false>(rn, Tj+E1, a.cW0,a.cb0,CB0,CB1,a.cb1,a.cW2,a.cb2, lane);

  // ---- per-edge scalar physics (D = 3) ----
  const float Aij = Ami*Amj, Bij = Bmi*Bmj, Cij = Cmi*Cmj;
  const float iE1 = 1000.f;   // 1/E1
  const float gAi = 2.f*Aij*(Amie*Amj - Aij)*iE1;
  const float gAj = 2.f*Aij*(Ami*Amje - Aij)*iE1;
  const float gBi = 2.f*Bij*(Bmie*Bmj - Bij)*iE1;
  const float gBj = 2.f*Bij*(Bmi*Bmje - Bij)*iE1;
  const float gCi = 2.f*Cij*(Cmie*Cmj - Cij)*iE1;
  const float gCj = 2.f*Cij*(Cmi*Cmje - Cij)*iE1;

  const float vix = a.v[i*3+0]-a.v[j*3+0];
  const float viy = a.v[i*3+1]-a.v[j*3+1];
  const float viz = a.v[i*3+2]-a.v[j*3+2];
  const float ev = ex*vix + ey*viy + ez*viz;
  const float vv = vix*vix + viy*viy + viz*viz;

  const float A2 = Aij*Aij, B2 = Bij*Bij;
  const float third = 1.f/3.f;
  const float coefM = 0.5f*A2 + (B2 - A2)*third;

  const float auxx = 0.5f*A2*vix + coefM*ev*ex;
  const float auxy = 0.5f*A2*viy + coefM*ev*ey;
  const float auxz = 0.5f*A2*viz + coefM*ev*ez;

  const float iTi = fast_rcp(Ti), iTj = fast_rcp(Tj);
  const float iCi = fast_rcp(Ci), iCj = fast_rcp(Cj);
  const float iCTi = iCi*iTi, iCTj = iCj*iTj;

  const float sT = iTi + iTj;
  const float msvx = sT*auxx, msvy = sT*auxy, msvz = sT*auxz;

  const float ct = -(iCTi + iCTj);
  const float cgi = (0.5f*gAi + (gBi - gAi)*third);
  const float cgj = (0.5f*gAj + (gBj - gAj)*third);
  const float dvx = ct*auxx + (0.5f*gAi*vix + cgi*ev*ex)*iCi + (0.5f*gAj*vix + cgj*ev*ex)*iCj;
  const float dvy = ct*auxy + (0.5f*gAi*viy + cgi*ev*ey)*iCi + (0.5f*gAj*viy + cgj*ev*ey)*iCj;
  const float dvz = ct*auxz + (0.5f*gAi*viz + cgi*ev*ez)*iCi + (0.5f*gAj*viz + cgj*ev*ez)*iCj;

  const float pcoef = Pi*fast_rcp(di*di) + Pj*fast_rcp(dj*dj);
  const float pdx = pcoef*gWx, pdy = pcoef*gWy, pdz = pcoef*gWz;

  const float auxMSS = (0.5f*A2*vv + coefM*ev*ev)*0.25f;
  const float Cij2 = Cij*Cij;
  const float mgss_i = sT*auxMSS + (iTi - iTj)*Cij2;
  const float mgss_j = sT*auxMSS + (iTj - iTi)*Cij2;

  const float hi   = (0.5f*gAi*vv + cgi*ev*ev)*iCi*0.25f;
  const float hj   = (0.5f*gAj*vv + cgj*ev*ev)*iCj*0.25f;
  const float hmix = (0.5f*gAi*vv + (0.5f*gAj + (gBi - gAi)*third)*ev*ev)*iCi*0.25f;

  const float inv_m = __expf(-a.logm[0]);
  const float tmv = -(2.f*A2 + (B2 - A2)*third);
  const float s5i = gCi*iCi - gCj*iCj;
  const float dms_i = -(2.f*iCTi + iCTj)*auxMSS + hi + hj
                      - (2.f*iCTi - iCTj)*Cij2 + s5i + tmv*inv_m;
  const float dms_j = -(2.f*iCTj + iCTi)*auxMSS + hj + hmix
                      - (2.f*iCTj - iCTi)*Cij2 - s5i + tmv*inv_m;

  // Noise: (A_ij * dW_bar + B_ij * I*tr/3) @ e_ij
  const float d00=a.dWn[ec*9+0], d01=a.dWn[ec*9+1], d02=a.dWn[ec*9+2];
  const float d10=a.dWn[ec*9+3], d11=a.dWn[ec*9+4], d12=a.dWn[ec*9+5];
  const float d20=a.dWn[ec*9+6], d21=a.dWn[ec*9+7], d22=a.dWn[ec*9+8];
  const float trd = (d00 + d11 + d22)*third;
  const float s01 = 0.5f*(d01 + d10), s02 = 0.5f*(d02 + d20), s12 = 0.5f*(d12 + d21);
  const float btr = Bij*trd;
  const float nx = (Aij*(d00 - trd) + btr)*ex + Aij*s01*ey + Aij*s02*ez;
  const float ny = Aij*s01*ex + (Aij*(d11 - trd) + btr)*ey + Aij*s12*ez;
  const float nz = Aij*s02*ex + Aij*s12*ey + (Aij*(d22 - trd) + btr)*ez;

  const float tn = -0.5f*(nx*vix + ny*viy + nz*viz);
  const float tC = Cij * a.dV[ec];

  if (valid && lane < 16) {
    atomicAdd(&a.accPd[i*3+0],  pdx); atomicAdd(&a.accPd[j*3+0], -pdx);
    atomicAdd(&a.accPd[i*3+1],  pdy); atomicAdd(&a.accPd[j*3+1], -pdy);
    atomicAdd(&a.accPd[i*3+2],  pdz); atomicAdd(&a.accPd[j*3+2], -pdz);

    atomicAdd(&a.accMSV[i*3+0],  msvx); atomicAdd(&a.accMSV[j*3+0], -msvx);
    atomicAdd(&a.accMSV[i*3+1],  msvy); atomicAdd(&a.accMSV[j*3+1], -msvy);
    atomicAdd(&a.accMSV[i*3+2],  msvz); atomicAdd(&a.accMSV[j*3+2], -msvz);

    atomicAdd(&a.accDivV[i*3+0],  dvx); atomicAdd(&a.accDivV[j*3+0], -dvx);
    atomicAdd(&a.accDivV[i*3+1],  dvy); atomicAdd(&a.accDivV[j*3+1], -dvy);
    atomicAdd(&a.accDivV[i*3+2],  dvz); atomicAdd(&a.accDivV[j*3+2], -dvz);

    atomicAdd(&a.accNoise[i*3+0],  nx); atomicAdd(&a.accNoise[j*3+0], -nx);
    atomicAdd(&a.accNoise[i*3+1],  ny); atomicAdd(&a.accNoise[j*3+1], -ny);
    atomicAdd(&a.accNoise[i*3+2],  nz); atomicAdd(&a.accNoise[j*3+2], -nz);

    atomicAdd(&a.accMgSS[i], mgss_i);  atomicAdd(&a.accMgSS[j], mgss_j);
    atomicAdd(&a.accDivMS[i], dms_i);  atomicAdd(&a.accDivMS[j], dms_j);
    atomicAdd(&a.accTnC[i], tn + tC);  atomicAdd(&a.accTnC[j], tn - tC);
  }
}

// ---------------------------------------------------------------------------
// Finalize: combine node accumulators into dvdt, dSdt, E_out.
// ---------------------------------------------------------------------------
__global__ void finalize_kernel(const float* __restrict__ v,
                                const float* __restrict__ Tn,
                                const float* __restrict__ Un,
                                const float* __restrict__ accPd,
                                const float* __restrict__ accMSV,
                                const float* __restrict__ accDivV,
                                const float* __restrict__ accNoise,
                                const float* __restrict__ accMgSS,
                                const float* __restrict__ accDivMS,
                                const float* __restrict__ accTnC,
                                const float* __restrict__ logkB,
                                const float* __restrict__ logm,
                                float* __restrict__ out, int N)
{
  const int n = blockIdx.x * blockDim.x + threadIdx.x;
  if (n >= N) return;
  const float kB = __expf(logkB[0]);
  const float m  = __expf(logm[0]);
  const float inv_m = __expf(-logm[0]);
  const float sdt = 10.f;                // 1/sqrt(DT), DT = 0.01
  const float s2k = sqrtf(2.f * kB);
  const float iT = fast_rcp(Tn[n]);
#pragma unroll
  for (int c = 0; c < 3; ++c) {
    const float dv = (-accPd[n*3+c] - 0.5f*accMSV[n*3+c] - 0.5f*kB*accDivV[n*3+c]) * inv_m
                   + s2k * inv_m * accNoise[n*3+c] * sdt;
    out[n*3+c] = dv;
  }
  out[3*N + n] = (accMgSS[n] + kB*accDivMS[n]) * iT + s2k * iT * accTnC[n] * sdt;
  const float vx = v[n*3+0], vy = v[n*3+1], vz = v[n*3+2];
  out[4*N + n] = Un[n] + 0.5f * m * (vx*vx + vy*vy + vz*vz);
}

// ---------------------------------------------------------------------------
// kernel_launch
// ---------------------------------------------------------------------------
extern "C" void kernel_launch(void* const* d_in, const int* in_sizes, int n_in,
                              void* d_out, int out_size, void* d_ws, size_t ws_size,
                              hipStream_t stream) {
  const int N = in_sizes[3];          // S has N elements
  const int E = in_sizes[6];          // dV has E elements

  const float* v   = (const float*)d_in[0];
  const int*   ei  = (const int*)d_in[1];
  const float* rij = (const float*)d_in[2];
  const float* S   = (const float*)d_in[3];
  const float* dAr = (const float*)d_in[4];
  const float* dW  = (const float*)d_in[5];
  const float* dV  = (const float*)d_in[6];
  // params leaves in jax pytree (sorted-key) order:
  // A_mlp(6), B_mlp(6), C_mlp(6), E_mlp(6), W_mlp(6), log_k_B, log_m
  const float* const* P = (const float* const*)(d_in + 7);

  float* wsf = (float*)d_ws;
  float* Tn = wsf;               float* Pn = wsf + (size_t)N;
  float* Cn = wsf + 2*(size_t)N; float* Un = wsf + 3*(size_t)N;
  float* accPd   = wsf + 4*(size_t)N;
  float* accMSV  = wsf + 7*(size_t)N;
  float* accDivV = wsf + 10*(size_t)N;
  float* accNoise= wsf + 13*(size_t)N;
  float* accMgSS = wsf + 16*(size_t)N;
  float* accDivMS= wsf + 17*(size_t)N;
  float* accTnC  = wsf + 18*(size_t)N;

  hipError_t err = hipMemsetAsync(accPd, 0, (size_t)15 * N * sizeof(float), stream);
  (void)err;

  const int ntiles = (N + TILE - 1) / TILE;
  const int nblocks = (ntiles + WAVES_PER_BLOCK - 1) / WAVES_PER_BLOCK;
  node_kernel<<<nblocks, 32 * WAVES_PER_BLOCK, 0, stream>>>(
      S, dAr,
      P[18], P[19], P[20], P[21], P[22], P[23],   // E_mlp
      Tn, Pn, Cn, Un, N);

  EdgeArgs ea;
  ea.v = v; ea.ei = ei; ea.rij = rij; ea.dArr = dAr; ea.dWn = dW; ea.dV = dV;
  ea.aW0 = P[0];  ea.ab0 = P[1];  ea.aW1 = P[2];  ea.ab1 = P[3];  ea.aW2 = P[4];  ea.ab2 = P[5];
  ea.bW0 = P[6];  ea.bb0 = P[7];  ea.bW1 = P[8];  ea.bb1 = P[9];  ea.bW2 = P[10]; ea.bb2 = P[11];
  ea.cW0 = P[12]; ea.cb0 = P[13]; ea.cW1 = P[14]; ea.cb1 = P[15]; ea.cW2 = P[16]; ea.cb2 = P[17];
  ea.wW0 = P[24]; ea.wb0 = P[25]; ea.wW1 = P[26]; ea.wb1 = P[27]; ea.wW2 = P[28]; ea.wb2 = P[29];
  ea.logm = P[31];
  ea.Tn = Tn; ea.Pn = Pn; ea.Cn = Cn;
  ea.accPd = accPd; ea.accMSV = accMSV; ea.accDivV = accDivV; ea.accNoise = accNoise;
  ea.accMgSS = accMgSS; ea.accDivMS = accDivMS; ea.accTnC = accTnC;
  ea.N = N; ea.E = E;

  const int etiles = (E + TILE - 1) / TILE;
  const int eblocks = (etiles + WAVES_PER_BLOCK - 1) / WAVES_PER_BLOCK;
  edge_kernel<<<eblocks, 32 * WAVES_PER_BLOCK, 0, stream>>>(ea);

  finalize_kernel<<<(N + 255) / 256, 256, 0, stream>>>(
      v, Tn, Un, accPd, accMSV, accDivV, accNoise, accMgSS, accDivMS, accTnC,
      P[30], P[31], (float*)d_out, N);
}
